// ImprovedSparseSimilarity_80135499809322
// MI455X (gfx1250) — matrix operand order repaired and checked by reading, not compiled, not run on hardware
//
#include <hip/hip_runtime.h>
#include <hip/hip_bf16.h>

// ---- problem constants (match reference setup_inputs) ----
#define BATCH 2
#define NROW  8192      // Nx == Ny
#define CDIM  512
#define KNBR  15
#define INV_TAU 5.0f    // 1/0.2

#define YROWS   64          // columns of sim handled per block iteration
#define KPH     128         // K per async phase (4 phases cover CDIM)
#define YPAD    136         // LDS row stride in halves (272B = 68 dwords -> conflict-free)
#define NPHASE  ((NROW / YROWS) * (CDIM / KPH))   // 512

typedef __attribute__((ext_vector_type(16))) __bf16 v16bf;
typedef __attribute__((ext_vector_type(8)))  __bf16 v8bf;
typedef __attribute__((ext_vector_type(8)))  float  v8f;

// RTNE float -> bf16
static __device__ __forceinline__ unsigned short f2bf(float f) {
    union { float f; unsigned u; } a; a.f = f;
    unsigned u = a.u;
    unsigned r = (u + 0x7FFFu + ((u >> 16) & 1u)) >> 16;
    return (unsigned short)r;
}

// Fully branchless sorted-descending top-15 insert: pure v_cndmask chains,
// arrays stay in registers (all indices compile-time after unroll).
#define TOPK_INSERT(tvA, tiA, vv, ii)                                       \
    do {                                                                    \
        _Pragma("unroll")                                                   \
        for (int _i = 14; _i > 0; --_i) {                                   \
            const bool _sh = (vv) > tvA[_i - 1];                            \
            const bool _pl = !_sh && ((vv) > tvA[_i]);                      \
            tvA[_i] = _sh ? tvA[_i - 1] : (_pl ? (vv) : tvA[_i]);           \
            tiA[_i] = _sh ? tiA[_i - 1] : (_pl ? (ii) : tiA[_i]);           \
        }                                                                   \
        const bool _c0 = (vv) > tvA[0];                                     \
        tvA[0] = _c0 ? (vv) : tvA[0];                                       \
        tiA[0] = _c0 ? (ii) : tiA[0];                                       \
    } while (0)

// ---------------------------------------------------------------------------
// Kernel 1: L2-normalize rows of X and Y (fp32) -> bf16 rows in workspace.
// One wave32 per row; 16 floats per lane; wave shuffle reduction.
// ---------------------------------------------------------------------------
__global__ __launch_bounds__(256) void norm_rows_bf16_kernel(
    const float* __restrict__ fx, const float* __restrict__ fy,
    unsigned short* __restrict__ xn, unsigned short* __restrict__ yn,
    int rows_per_src /* BATCH*NROW */)
{
    const int waveg = (blockIdx.x * blockDim.x + threadIdx.x) >> 5;
    const int lane  = threadIdx.x & 31;

    const float* src;
    unsigned short* dst;
    int row;
    if (waveg < rows_per_src) { src = fx; dst = xn; row = waveg; }
    else                      { src = fy; dst = yn; row = waveg - rows_per_src; }

    const float4* p4 = (const float4*)(src + (size_t)row * CDIM);

    float4 v[4];
    float ss = 0.0f;
#pragma unroll
    for (int c = 0; c < 4; ++c) {
        v[c] = p4[lane + c * 32];
        ss += v[c].x * v[c].x + v[c].y * v[c].y + v[c].z * v[c].z + v[c].w * v[c].w;
    }
#pragma unroll
    for (int m = 16; m > 0; m >>= 1) ss += __shfl_xor(ss, m, 32);

    const float inv = 1.0f / fmaxf(sqrtf(ss), 1e-12f);

    uint2* o2 = (uint2*)(dst + (size_t)row * CDIM);
#pragma unroll
    for (int c = 0; c < 4; ++c) {
        uint2 o;
        o.x = (unsigned)f2bf(v[c].x * inv) | ((unsigned)f2bf(v[c].y * inv) << 16);
        o.y = (unsigned)f2bf(v[c].z * inv) | ((unsigned)f2bf(v[c].w * inv) << 16);
        o2[lane + c * 32] = o;
    }
}

// ---------------------------------------------------------------------------
// Kernel 2: fused  sim = (Xn Yn^T)/tau -> streaming per-row top-15 -> softmax.
// Block = 256 thr = 8 waves, owns 32 rows of one batch.
// Y panel (64 cols x 128 K) is double-buffered in LDS via
// global_load_async_to_lds_b128 (ASYNCcnt pipeline); WMMA B-fragments come
// from LDS (padded stride, conflict-free), A-fragments from global (WGP$-hot).
// ---------------------------------------------------------------------------
__global__ __launch_bounds__(256) void sim_topk_softmax_kernel(
    const __bf16* __restrict__ Xn, const __bf16* __restrict__ Yn,
    float* __restrict__ out_vals, int* __restrict__ out_idx)
{
    __shared__ __align__(16) unsigned short Ybuf[2][YROWS][YPAD]; // 2 x 17408 B
    __shared__ float lds_sim[32][65];                             // 8320 B

    const int blocksPerBatch = NROW / 32;
    const int batch = blockIdx.x / blocksPerBatch;
    const int m0    = (blockIdx.x % blocksPerBatch) * 32;

    const __bf16* X = Xn + (size_t)batch * NROW * CDIM;
    const __bf16* Y = Yn + (size_t)batch * NROW * CDIM;

    const int tid   = threadIdx.x;
    const int wave  = tid >> 5;
    const int lane  = tid & 31;
    const int wRow  = (wave >> 2) * 16;    // 0 | 16   (row sub-tile)
    const int wCol  = (wave & 3) * 16;     // 0..48    (col sub-tile)
    const int half  = lane >> 4;           // half-wave select
    const int l15   = lane & 15;

    // A fragment (16x32 bf16, ISA 7.12.2): lane -> M = l15,
    // halves [0..7] = K(half?8:0)+{0..7}, halves [8..15] = +16.
    const __bf16* arow = X + (size_t)(m0 + wRow + l15) * CDIM + half * 8;

    // B fragment row base in LDS: lane -> N = wCol + l15
    const unsigned ybase[2] = { (unsigned)(uintptr_t)&Ybuf[0][0][0],
                                (unsigned)(uintptr_t)&Ybuf[1][0][0] };

    // merge-state partitioning: 8 threads per row
    const int mrow = tid >> 3;             // 0..31
    const int msub = tid & 7;

    float tv[KNBR]; int ti[KNBR];
#pragma unroll
    for (int i = 0; i < KNBR; ++i) { tv[i] = -3.402823466e38f; ti[i] = 0; }

    // --- async panel copy: 64 rows x 256 B = 16 KB = 1024 x 16B chunks;
    //     each of 256 threads issues 4 global_load_async_to_lds_b128.
    auto issue_panel = [&](int j0, int kp, unsigned ldsBase) {
#pragma unroll
        for (int i = 0; i < 4; ++i) {
            const int chunk = tid + i * 256;
            const int r = chunk >> 4;        // panel row 0..63
            const int q = chunk & 15;        // 16B chunk within 256B row slice
            const unsigned long long src =
                (unsigned long long)(uintptr_t)(Y + (size_t)(j0 + r) * CDIM + kp * KPH + q * 8);
            const unsigned dst = ldsBase + (unsigned)(r * (YPAD * 2) + q * 16);
            asm volatile("global_load_async_to_lds_b128 %0, %1, off"
                         :: "v"(dst), "v"(src) : "memory");
        }
    };

    int buf = 0;
    issue_panel(0, 0, ybase[0]);

    v8f acc = {};
    for (int t = 0; t < NPHASE; ++t) {
        const int j0 = (t >> 2) * YROWS;
        const int kp = t & 3;

        // prefetch next phase's panel into the other buffer, then wait for ours
        if (t + 1 < NPHASE) {
            const int nt = t + 1;
            issue_panel((nt >> 2) * YROWS, nt & 3, ybase[buf ^ 1]);
            asm volatile("s_wait_asynccnt 0x4" ::: "memory");   // our panel done
        } else {
            asm volatile("s_wait_asynccnt 0x0" ::: "memory");
        }
        __syncthreads();                                        // panel ready WG-wide

        if (kp == 0) acc = (v8f){};

        const __bf16* yrow = (const __bf16*)&Ybuf[buf][wCol + l15][0] + half * 8;
        const __bf16* arp  = arow + kp * KPH;
#pragma unroll
        for (int kk = 0; kk < KPH; kk += 32) {
            v8bf a0 = *(const v8bf*)(arp + kk);
            v8bf a1 = *(const v8bf*)(arp + kk + 16);
            v8bf b0 = *(const v8bf*)(yrow + kk);       // ds_load_b128, stride 68 dw
            v8bf b1 = *(const v8bf*)(yrow + kk + 16);
            v16bf af  = __builtin_shufflevector(a0, a1, 0,1,2,3,4,5,6,7,8,9,10,11,12,13,14,15);
            v16bf bfv = __builtin_shufflevector(b0, b1, 0,1,2,3,4,5,6,7,8,9,10,11,12,13,14,15);
            acc = __builtin_amdgcn_wmma_f32_16x16x32_bf16(
                false, af, false, bfv, (short)0, acc, false, false);
        }

        if (kp == 3) {
            // C/D layout: VGPR r, lane -> M = r + 8*half, N = l15
#pragma unroll
            for (int r = 0; r < 8; ++r)
                lds_sim[wRow + r + half * 8][wCol + l15] = acc[r] * INV_TAU;
            __syncthreads();            // sim tile visible + panel reads finished

            float thr = tv[KNBR - 1];
#pragma unroll
            for (int c8 = 0; c8 < 8; ++c8) {
                const int c = msub + c8 * 8;
                const float vv = lds_sim[mrow][c];
                if (vv > thr) {                       // rare: branchless insert
                    TOPK_INSERT(tv, ti, vv, j0 + c);
                    thr = tv[KNBR - 1];
                }
            }
            __syncthreads();            // sim consumed; safe for next overwrite
        } else {
            __syncthreads();            // all waves done reading Ybuf[buf]
        }
        buf ^= 1;
    }

    // ---- final 8-way merge per row; candidate arrays alias the Y buffers ----
    float* cand_v = (float*)&Ybuf[0][0][0];   // 32*120 floats = 15360 B < 17408
    int*   cand_i = (int*)&Ybuf[1][0][0];
#pragma unroll
    for (int i = 0; i < KNBR; ++i) {
        cand_v[mrow * (8 * KNBR) + msub * KNBR + i] = tv[i];
        cand_i[mrow * (8 * KNBR) + msub * KNBR + i] = ti[i];
    }
    __syncthreads();

    if (msub == 0) {
        float fv[KNBR]; int fi[KNBR];
#pragma unroll
        for (int i = 0; i < KNBR; ++i) { fv[i] = -3.402823466e38f; fi[i] = 0; }
        for (int c = 0; c < 8 * KNBR; ++c) {
            const float vv = cand_v[mrow * (8 * KNBR) + c];
            if (vv > fv[KNBR - 1])
                TOPK_INSERT(fv, fi, vv, cand_i[mrow * (8 * KNBR) + c]);
        }
        // softmax over the (descending-sorted) top-15 — matches lax.top_k order
        const float mx = fv[0];
        float e[KNBR], s = 0.0f;
#pragma unroll
        for (int i = 0; i < KNBR; ++i) { e[i] = __expf(fv[i] - mx); s += e[i]; }
        const float invs = 1.0f / s;
        const size_t g = ((size_t)batch * NROW + (m0 + mrow)) * KNBR;
#pragma unroll
        for (int i = 0; i < KNBR; ++i) {
            out_vals[g + i] = e[i] * invs;
            out_idx[g + i]  = fi[i];
        }
    }
}

// ---------------------------------------------------------------------------
extern "C" void kernel_launch(void* const* d_in, const int* in_sizes, int n_in,
                              void* d_out, int out_size, void* d_ws, size_t ws_size,
                              hipStream_t stream) {
    (void)in_sizes; (void)n_in; (void)out_size; (void)ws_size;

    const float* fx = (const float*)d_in[0];   // [2,8192,512] fp32
    const float* fy = (const float*)d_in[1];   // [2,8192,512] fp32

    // workspace: normalized bf16 copies (16 MB each)
    unsigned short* xn = (unsigned short*)d_ws;
    unsigned short* yn = xn + (size_t)BATCH * NROW * CDIM;

    const int rowsPerSrc = BATCH * NROW;                 // 16384
    const int normBlocks = (2 * rowsPerSrc) / 8;         // 8 waves/block
    norm_rows_bf16_kernel<<<normBlocks, 256, 0, stream>>>(fx, fy, xn, yn, rowsPerSrc);

    // outputs: [2,8192,15] f32 softmax, then [2,8192,15] i32 indices
    float* out_vals = (float*)d_out;
    int*   out_idx  = (int*)((float*)d_out + (size_t)BATCH * NROW * KNBR);

    const int gemmBlocks = BATCH * (NROW / 32);          // 512 blocks
    sim_topk_softmax_kernel<<<gemmBlocks, 256, 0, stream>>>(
        (const __bf16*)xn, (const __bf16*)yn, out_vals, out_idx);
}
